// NeuTraj_Network_85177791414548
// MI455X (gfx1250) — compile-verified, matrix operand order checked by reading, not run on hardware
//
#include <hip/hip_runtime.h>

#define BSZ  220
#define TLEN 512
#define HID  128
#define G4H  512
#define SAMP 10
#define MT   14   // ceil(220/16)

typedef __attribute__((ext_vector_type(16))) __bf16 v16bf;
typedef __attribute__((ext_vector_type(8)))  float  v8f;

__device__ __forceinline__ unsigned short f2bf_bits(float f) {
  union { float f; unsigned u; } v; v.f = f;
  unsigned r = v.u + 0x7FFFu + ((v.u >> 16) & 1u);   // RNE
  return (unsigned short)(r >> 16);
}
__device__ __forceinline__ __bf16 bits2bf(unsigned short h) {
  __bf16 b; __builtin_memcpy(&b, &h, 2); return b;
}
__device__ __forceinline__ float sigf(float x) { return 1.0f / (1.0f + __expf(-x)); }

// 16-bit A-matrix 16x32 per-lane element->K mapping (ISA 7.12.2); B uses the
// symmetric column-major layout (lane = N, same K pattern).
__device__ __forceinline__ int kmap(int e, int lane) {
  int v = e >> 1, p = e & 1;
  return ((v >> 2) << 4) + (((lane >> 4) & 1) << 3) + ((v & 3) << 1) + p;
}

__global__ __launch_bounds__(512)
void lstm_gather_kernel(const float* __restrict__ x_t, const float* __restrict__ x_a,
                        const float* __restrict__ x_f,
                        const int* __restrict__ len_t, const int* __restrict__ len_a,
                        const int* __restrict__ len_f,
                        const int* __restrict__ tasl, const int* __restrict__ tfsl,
                        const int* __restrict__ asl,  const int* __restrict__ fsl,
                        const float* __restrict__ W_ih, const float* __restrict__ W_hh,
                        const float* __restrict__ b_ih, const float* __restrict__ b_hh,
                        float* __restrict__ emb, float* __restrict__ ta, float* __restrict__ tf,
                        float* __restrict__ an, float* __restrict__ fa) {
  __shared__ float          g_buf[16][G4H];   // 32 KB gate pre-activations
  __shared__ float          c_buf[16][HID];   //  8 KB cell state
  __shared__ unsigned short h_bits[16][HID];  //  4 KB hidden state (bf16)
  __shared__ float          wi0[G4H], wi1[G4H], bsum[G4H]; // 6 KB input proj
  __shared__ int            need_t[16][21];

  const int tid  = threadIdx.x;
  const int lane = tid & 31;
  const int wid  = tid >> 5;          // 16 waves
  const int seq  = blockIdx.x / MT;
  const int mt   = blockIdx.x % MT;

  const float* x; const int* lenA; const int* subA; const int* subB;
  float* sub1; float* sub2;
  if (seq == 0)      { x = x_t; lenA = len_t; subA = tasl; subB = tfsl; sub1 = ta; sub2 = tf; }
  else if (seq == 1) { x = x_a; lenA = len_a; subA = asl;  subB = asl;  sub1 = an; sub2 = an; }
  else               { x = x_f; lenA = len_f; subA = fsl;  subB = fsl;  sub1 = fa; sub2 = fa; }
  float* embp = emb + (size_t)seq * BSZ * HID;

  // ---- one-time LDS init ----
  if (tid < G4H) {
    wi0[tid]  = W_ih[tid * 2 + 0];
    wi1[tid]  = W_ih[tid * 2 + 1];
    bsum[tid] = b_ih[tid] + b_hh[tid];
  }
  {
    int m = tid >> 5, j0 = (tid & 31) * 4;
    #pragma unroll
    for (int q = 0; q < 4; ++q) { c_buf[m][j0 + q] = 0.f; h_bits[m][j0 + q] = 0; }
  }
  if (tid < 16) {
    int b = mt * 16 + tid;
    need_t[tid][0] = (b < BSZ) ? ((lenA[b] - 1 > 0) ? lenA[b] - 1 : 0) : -1;
    for (int s = 0; s < SAMP; ++s) {
      int ka = (b < BSZ) ? subA[b * SAMP + s] - 1 : -1;
      int kb = (b < BSZ && seq == 0) ? subB[b * SAMP + s] - 1 : -1;
      need_t[tid][1 + s]  = (b < BSZ) ? (ka > 0 ? ka : 0) : -1;
      need_t[tid][11 + s] = (b < BSZ && seq == 0) ? (kb > 0 ? kb : 0) : -1;
    }
  }

  // ---- W_hh^T bf16 B-fragments resident in registers (reused 512 steps) ----
  v16bf bfrag[2][4];
  #pragma unroll
  for (int n2 = 0; n2 < 2; ++n2) {
    const int gate = (wid * 2 + n2) * 16 + (lane & 15);
    #pragma unroll
    for (int kc = 0; kc < 4; ++kc)
      #pragma unroll
      for (int e = 0; e < 16; ++e)
        bfrag[n2][kc][e] = bits2bf(f2bf_bits(W_hh[gate * HID + kc * 32 + kmap(e, lane)]));
  }
  __syncthreads();

  const int em = tid >> 5;          // elementwise row 0..15
  const int ej = (tid & 31) * 4;    // elementwise hidden base
  const int eb = mt * 16 + em;

  int nT0, nTa[SAMP], nTb[SAMP];
  nT0 = need_t[em][0];
  #pragma unroll
  for (int s = 0; s < SAMP; ++s) { nTa[s] = need_t[em][1 + s]; nTb[s] = need_t[em][11 + s]; }

  for (int t = 0; t < TLEN; ++t) {
    // A-fragments from resident hidden state
    v16bf afrag[4];
    #pragma unroll
    for (int kc = 0; kc < 4; ++kc)
      #pragma unroll
      for (int e = 0; e < 16; ++e)
        afrag[kc][e] = bits2bf(h_bits[lane & 15][kc * 32 + kmap(e, lane)]);

    v8f acc0 = {0.f,0.f,0.f,0.f,0.f,0.f,0.f,0.f};
    v8f acc1 = {0.f,0.f,0.f,0.f,0.f,0.f,0.f,0.f};
    #pragma unroll
    for (int kc = 0; kc < 4; ++kc) {
      acc0 = __builtin_amdgcn_wmma_f32_16x16x32_bf16(false, afrag[kc], false, bfrag[0][kc],
                                                     (short)0, acc0, false, false);
      acc1 = __builtin_amdgcn_wmma_f32_16x16x32_bf16(false, afrag[kc], false, bfrag[1][kc],
                                                     (short)0, acc1, false, false);
    }
    { // scatter D tiles to g_buf (C/D layout: VGPR r -> M = r + 8*(lane>=16), N = lane&15)
      const int n0 = (wid * 2 + 0) * 16 + (lane & 15);
      const int n1 = (wid * 2 + 1) * 16 + (lane & 15);
      const int mo = (lane >> 4) << 3;
      #pragma unroll
      for (int r = 0; r < 8; ++r) {
        g_buf[mo + r][n0] = acc0[r];
        g_buf[mo + r][n1] = acc1[r];
      }
    }
    __syncthreads();

    // ---- fused input-projection + gates + state update + inline gather ----
    {
      float x0 = 0.f, x1 = 0.f;
      if (eb < BSZ) {
        const float* xp = x + ((size_t)eb * TLEN + t) * 4;
        x0 = xp[0]; x1 = xp[1];
      }
      float hv[4];
      #pragma unroll
      for (int q = 0; q < 4; ++q) {
        const int jh = ej + q;
        float pi = g_buf[em][jh]           + x0 * wi0[jh]           + x1 * wi1[jh]           + bsum[jh];
        float pf = g_buf[em][HID + jh]     + x0 * wi0[HID + jh]     + x1 * wi1[HID + jh]     + bsum[HID + jh];
        float pg = g_buf[em][2 * HID + jh] + x0 * wi0[2 * HID + jh] + x1 * wi1[2 * HID + jh] + bsum[2 * HID + jh];
        float po = g_buf[em][3 * HID + jh] + x0 * wi0[3 * HID + jh] + x1 * wi1[3 * HID + jh] + bsum[3 * HID + jh];
        float c  = sigf(pf) * c_buf[em][jh] + sigf(pi) * tanhf(pg);
        c_buf[em][jh]  = c;
        float h        = sigf(po) * tanhf(c);
        h_bits[em][jh] = f2bf_bits(h);
        hv[q] = h;
      }
      if (eb < BSZ) {
        if (t == nT0) {
          #pragma unroll
          for (int q = 0; q < 4; ++q) embp[(size_t)eb * HID + ej + q] = hv[q];
        }
        #pragma unroll 1
        for (int s = 0; s < SAMP; ++s) {
          const size_t k = (size_t)(eb * SAMP + s);
          if (t == nTa[s]) {
            #pragma unroll
            for (int q = 0; q < 4; ++q) sub1[k * HID + ej + q] = hv[q];
          }
          if (seq == 0 && t == nTb[s]) {
            #pragma unroll
            for (int q = 0; q < 4; ++q) sub2[k * HID + ej + q] = hv[q];
          }
        }
      }
    }
    __syncthreads();
  }
}

#define NPAIR 4840
__global__ __launch_bounds__(256)
void dist_kernel(const float* __restrict__ emb, const float* __restrict__ ta,
                 const float* __restrict__ tf, const float* __restrict__ an,
                 const float* __restrict__ fa, float* __restrict__ out) {
  const int lane = threadIdx.x & 31;
  const int wid  = threadIdx.x >> 5;
  const int p    = blockIdx.x * 8 + wid;   // one wave per distance
  if (p >= NPAIR) return;
  const float *pa, *pb;
  if (p < 220)       { pa = emb + (size_t)p * HID;            pb = emb + (size_t)(220 + p) * HID; }
  else if (p < 440)  { int b = p - 220;  pa = emb + (size_t)b * HID; pb = emb + (size_t)(440 + b) * HID; }
  else if (p < 2640) { int k = p - 440;  pa = ta + (size_t)k * HID;  pb = an + (size_t)k * HID; }
  else               { int k = p - 2640; pa = tf + (size_t)k * HID;  pb = fa + (size_t)k * HID; }
  float s = 0.f;
  #pragma unroll
  for (int j = lane; j < HID; j += 32) { float d = pa[j] - pb[j]; s += d * d; }
  #pragma unroll
  for (int off = 16; off > 0; off >>= 1) s += __shfl_down(s, off, 32);
  if (lane == 0) out[p] = expf(-sqrtf(s));
}

extern "C" void kernel_launch(void* const* d_in, const int* in_sizes, int n_in,
                              void* d_out, int out_size, void* d_ws, size_t ws_size,
                              hipStream_t stream) {
  const float* x_t   = (const float*)d_in[0];
  const int*   len_t = (const int*)  d_in[1];
  const int*   tasl  = (const int*)  d_in[2];
  const int*   tfsl  = (const int*)  d_in[3];
  const float* x_a   = (const float*)d_in[4];
  const int*   len_a = (const int*)  d_in[5];
  const int*   asl   = (const int*)  d_in[6];
  const float* x_f   = (const float*)d_in[7];
  const int*   len_f = (const int*)  d_in[8];
  const int*   fsl   = (const int*)  d_in[9];
  const float* W_ih  = (const float*)d_in[10];
  const float* W_hh  = (const float*)d_in[11];
  const float* b_ih  = (const float*)d_in[12];
  const float* b_hh  = (const float*)d_in[13];

  float* ws  = (float*)d_ws;
  float* emb = ws;                          // [3*220][128]
  float* ta  = emb + (size_t)3 * BSZ * HID; // [2200][128] each
  float* tf  = ta  + (size_t)BSZ * SAMP * HID;
  float* an  = tf  + (size_t)BSZ * SAMP * HID;
  float* fa  = an  + (size_t)BSZ * SAMP * HID;

  lstm_gather_kernel<<<3 * MT, 512, 0, stream>>>(x_t, x_a, x_f, len_t, len_a, len_f,
                                                 tasl, tfsl, asl, fsl,
                                                 W_ih, W_hh, b_ih, b_hh,
                                                 emb, ta, tf, an, fa);
  dist_kernel<<<(NPAIR + 7) / 8, 256, 0, stream>>>(emb, ta, tf, an, fa, (float*)d_out);
}
